// ComplexRNN_63488206569508
// MI455X (gfx1250) — matrix operand order repaired and checked by reading, not compiled, not run on hardware
//
#include <hip/hip_runtime.h>
#include <math.h>

typedef __attribute__((ext_vector_type(16))) __bf16 v16bf;
typedef __attribute__((ext_vector_type(8)))  float  v8f;
typedef __attribute__((ext_vector_type(8)))  int    v8i;
typedef __attribute__((ext_vector_type(4)))  int    v4i;
typedef __attribute__((ext_vector_type(2)))  float  f32x2;

#define HC   256      // complex hidden units
#define N2   512      // 2*HC real channels
#define TLEN 4096
#define BSZ  8

// ---------- helpers ----------
static __device__ __forceinline__ unsigned bf16rne(float f) {
  unsigned u = __float_as_uint(f);
  u += 0x7FFFu + ((u >> 16) & 1u);      // round-to-nearest-even
  return u >> 16;
}
static __device__ __forceinline__ unsigned packbf2(float a, float b) {
  return (bf16rne(a) & 0xFFFFu) | (bf16rne(b) << 16);
}
static __device__ __forceinline__ float tanh_fast(float x) {
#if __has_builtin(__builtin_amdgcn_tanhf)
  return __builtin_amdgcn_tanhf(x);
#elif __has_builtin(__builtin_amdgcn_tanh_f32)
  return __builtin_amdgcn_tanh_f32(x);
#else
  return tanhf(x);
#endif
}
// Combined complex matrix Cc[k][n] (512x512) from wr/wi (256x256):
//   [[Wr, Wi], [-Wi, Wr]]
static __device__ __forceinline__ float cval(const float* __restrict__ wr,
                                             const float* __restrict__ wi,
                                             int k, int n) {
  if (k < HC) return (n < HC) ? wr[k*HC + n] : wi[k*HC + (n - HC)];
  int kk = k - HC;
  return (n < HC) ? -wi[kk*HC + n] : wr[kk*HC + (n - HC)];
}
static __device__ __forceinline__ v8f wmma_bf16(v8i a, v8i b, v8f c) {
  return __builtin_amdgcn_wmma_f32_16x16x32_bf16(
      false, __builtin_bit_cast(v16bf, a),
      false, __builtin_bit_cast(v16bf, b),
      (short)0, c, false, false);
}

// ---------- prep: bake combined matrix into WMMA-B fragment order ----------
// dst dword index: ((j*16 + s)*32 + lane)*8 + d   (j = N-tile, s = K-step)
// B-frag (32x16 bf16): n = j*16 + (lane&15); K = s*32 + (d&4 ? 16 : 0) + (d&3)*2 + (lane>>4)*8 + {0,1}
__global__ void prep_frag(const float* __restrict__ wr,
                          const float* __restrict__ wi,
                          unsigned* __restrict__ dst) {
  int idx  = blockIdx.x * 256 + threadIdx.x;   // 131072 dwords total
  int d    = idx & 7;
  int lane = (idx >> 3) & 31;
  int s    = (idx >> 8) & 15;
  int j    = idx >> 12;
  int n = j*16 + (lane & 15);
  int k = s*32 + ((d & 4) ? 16 : 0) + (d & 3)*2 + (lane >> 4)*8;
  dst[idx] = packbf2(cval(wr, wi, k, n), cval(wr, wi, k + 1, n));
}

// ---------- input projection: wx = X @ Wc + (wb + ub) ----------
// in : [B*T, 512] (row = b*4096+t), wx out layout: [t][32 ntile][8 b][16 n]
__global__ __launch_bounds__(256) void proj_gemm(
    const float* __restrict__ in,
    const unsigned* __restrict__ wfrag,
    const float* __restrict__ wbr, const float* __restrict__ wbi,
    const float* __restrict__ ubr, const float* __restrict__ ubi,
    float* __restrict__ wx)
{
  int wave = threadIdx.x >> 5;
  int lane = threadIdx.x & 31;
  int half = lane >> 4;
  int r    = lane & 15;
  int mtile = blockIdx.x;                       // 0..2047
  const float* arow = in + (size_t)(mtile*16 + r) * N2;
  int j0 = wave * 4;                            // this wave's 4 N-tiles

  v8f acc[4] = {v8f{}, v8f{}, v8f{}, v8f{}};
  const unsigned* bbase = wfrag + lane*8;

  for (int s = 0; s < 16; ++s) {
    // issue ALL loads of this iteration before any pack/WMMA
    f32x2 xr[8];
    #pragma unroll
    for (int d = 0; d < 8; ++d) {
      int k = s*32 + ((d & 4) ? 16 : 0) + (d & 3)*2 + half*8;
      xr[d] = *(const f32x2*)(arow + k);
    }
    v8i bv[4];
    #pragma unroll
    for (int jj = 0; jj < 4; ++jj)
      bv[jj] = *(const v8i*)(bbase + (((j0 + jj)*16 + s) << 8));

    v8i av;
    #pragma unroll
    for (int d = 0; d < 8; ++d) av[d] = packbf2(xr[d][0], xr[d][1]);

    #pragma unroll
    for (int jj = 0; jj < 4; ++jj)
      acc[jj] = wmma_bf16(av, bv[jj], acc[jj]);
  }

  #pragma unroll
  for (int jj = 0; jj < 4; ++jj) {
    int n = (j0 + jj)*16 + r;
    float bias = (n < HC) ? (wbr[n] + ubr[n]) : (wbi[n-HC] + ubi[n-HC]);
    #pragma unroll
    for (int v = 0; v < 8; ++v) {
      int row = mtile*16 + v + half*8;          // D row = v + 8*half
      int b = row >> 12;                        // /4096
      int t = row & (TLEN - 1);
      wx[((t*32 + (j0 + jj))*8 + b)*16 + r] = acc[jj][v] + bias;
    }
  }
}

// ---------- recurrent scan: h_t = tanh(wx_t + h_{t-1} @ Uc) ----------
// One persistent 512-thread workgroup, 16 waves x 2 N-tiles.
// U fragments for K-steps 0..7 live permanently in VGPRs (time-invariant);
// K-steps 8..15 stream from L2 with a 1-deep software pipeline.
#define HROW 260   // dwords per LDS h row (260 % 64 == 4 -> conflict-free b128)

__global__ __launch_bounds__(512) void scan_rnn(
    const unsigned* __restrict__ ufrag,
    const float* __restrict__ wx,       // [t][32][8][16]
    float* __restrict__ hout)           // [8][4096][512]
{
  __shared__ unsigned hbuf[2][16][HROW];
  int wave = threadIdx.x >> 5;          // 0..15
  int lane = threadIdx.x & 31;
  int half = lane >> 4;
  int r    = lane & 15;
  int j0 = wave*2, j1 = wave*2 + 1;     // two N-tiles per wave

  for (int i = threadIdx.x; i < 2*16*HROW; i += 512)
    ((unsigned*)hbuf)[i] = 0u;          // h0 = 0, pad rows 8..15 stay 0
  __syncthreads();

  const unsigned* b0 = ufrag + j0*4096 + lane*8;
  const unsigned* b1 = ufrag + j1*4096 + lane*8;

  // resident U fragments: s = 0..7, both tiles (128 VGPRs)
  v8i u0[8], u1[8];
  #pragma unroll
  for (int s = 0; s < 8; ++s) {
    u0[s] = *(const v8i*)(b0 + s*256);
    u1[s] = *(const v8i*)(b1 + s*256);
  }

  for (int t = 0; t < TLEN; ++t) {
    const unsigned* hr = &hbuf[t & 1][r][half*4];
    unsigned short* hw = (unsigned short*)hbuf[(t & 1) ^ 1];

    // issue wx loads early; they retire behind the WMMA chain
    const float* wxp0 = wx + (size_t)(t*32 + j0)*128;
    const float* wxp1 = wx + (size_t)(t*32 + j1)*128;
    float wxv0[8], wxv1[8];
    #pragma unroll
    for (int v = 0; v < 8; ++v) {
      wxv0[v] = wxp0[v*16 + r];
      wxv1[v] = wxp1[v*16 + r];
    }

    v8f acc0 = {}, acc1 = {};

    // K-steps 0..7: weights already in registers, only LDS A loads
    #pragma unroll
    for (int s = 0; s < 8; ++s) {
      v4i alo = *(const v4i*)(hr + s*16);
      v4i ahi = *(const v4i*)(hr + s*16 + 8);
      v8i ai = (v8i){alo[0], alo[1], alo[2], alo[3],
                     ahi[0], ahi[1], ahi[2], ahi[3]};
      acc0 = wmma_bf16(ai, u0[s], acc0);
      acc1 = wmma_bf16(ai, u1[s], acc1);
    }

    // K-steps 8..15: stream B from L2, prefetch one step ahead
    v8i p0 = *(const v8i*)(b0 + 8*256);
    v8i p1 = *(const v8i*)(b1 + 8*256);
    #pragma unroll
    for (int s = 8; s < 16; ++s) {
      v8i c0 = p0, c1 = p1;
      if (s < 15) {
        p0 = *(const v8i*)(b0 + (s + 1)*256);
        p1 = *(const v8i*)(b1 + (s + 1)*256);
      }
      v4i alo = *(const v4i*)(hr + s*16);
      v4i ahi = *(const v4i*)(hr + s*16 + 8);
      v8i ai = (v8i){alo[0], alo[1], alo[2], alo[3],
                     ahi[0], ahi[1], ahi[2], ahi[3]};
      acc0 = wmma_bf16(ai, c0, acc0);
      acc1 = wmma_bf16(ai, c1, acc1);
    }

    // epilogue: add wx, tanh, publish h_t (rows 0..7 live in half==0 lanes)
    #pragma unroll
    for (int v = 0; v < 8; ++v) {
      float h0v = tanh_fast(acc0[v] + wxv0[v]);
      float h1v = tanh_fast(acc1[v] + wxv1[v]);
      if (half == 0) {
        hw[v*(2*HROW) + j0*16 + r] = (unsigned short)bf16rne(h0v);
        hw[v*(2*HROW) + j1*16 + r] = (unsigned short)bf16rne(h1v);
        hout[((size_t)v*TLEN + t)*N2 + j0*16 + r] = h0v;
        hout[((size_t)v*TLEN + t)*N2 + j1*16 + r] = h1v;
      }
    }
    __syncthreads();                    // publish h_t before next step
  }
}

// ---------- host launcher ----------
extern "C" void kernel_launch(void* const* d_in, const int* in_sizes, int n_in,
                              void* d_out, int out_size, void* d_ws, size_t ws_size,
                              hipStream_t stream) {
  (void)in_sizes; (void)n_in; (void)out_size; (void)ws_size;
  const float* x    = (const float*)d_in[0];
  const float* wr0  = (const float*)d_in[1];
  const float* wi0  = (const float*)d_in[2];
  const float* wbr0 = (const float*)d_in[3];
  const float* wbi0 = (const float*)d_in[4];
  const float* ur0  = (const float*)d_in[5];
  const float* ui0  = (const float*)d_in[6];
  const float* ubr0 = (const float*)d_in[7];
  const float* ubi0 = (const float*)d_in[8];
  const float* wr1  = (const float*)d_in[9];
  const float* wi1  = (const float*)d_in[10];
  const float* wbr1 = (const float*)d_in[11];
  const float* wbi1 = (const float*)d_in[12];
  const float* ur1  = (const float*)d_in[13];
  const float* ui1  = (const float*)d_in[14];
  const float* ubr1 = (const float*)d_in[15];
  const float* ubi1 = (const float*)d_in[16];

  char* ws = (char*)d_ws;
  unsigned* wc0 = (unsigned*)(ws);                       // 512 KB each
  unsigned* uc0 = (unsigned*)(ws + (512u << 10));
  unsigned* wc1 = (unsigned*)(ws + (1024u << 10));
  unsigned* uc1 = (unsigned*)(ws + (1536u << 10));
  float*    wx  = (float*)   (ws + (2048u << 10));       // 64 MB
  float*    out = (float*)d_out;

  prep_frag<<<512, 256, 0, stream>>>(wr0, wi0, wc0);
  prep_frag<<<512, 256, 0, stream>>>(ur0, ui0, uc0);
  prep_frag<<<512, 256, 0, stream>>>(wr1, wi1, wc1);
  prep_frag<<<512, 256, 0, stream>>>(ur1, ui1, uc1);

  // layer 0
  proj_gemm<<<2048, 256, 0, stream>>>(x,   wc0, wbr0, wbi0, ubr0, ubi0, wx);
  scan_rnn <<<1, 512, 0, stream>>>(uc0, wx, out);        // h0 -> d_out (temp)
  // layer 1 (reads layer-0 h from d_out, then overwrites d_out with final h)
  proj_gemm<<<2048, 256, 0, stream>>>(out, wc1, wbr1, wbi1, ubr1, ubi1, wx);
  scan_rnn <<<1, 512, 0, stream>>>(uc1, wx, out);
}
